// VanillaGNN_5600637354091
// MI455X (gfx1250) — matrix operand order repaired and checked by reading, not compiled, not run on hardware
//
#include <hip/hip_runtime.h>
#include <stdint.h>

#define N_NODES 100000
#define N_TILES (N_NODES / 16)   // 6250, exact

typedef __attribute__((ext_vector_type(16))) __bf16 v16bf;
typedef __attribute__((ext_vector_type(2)))  __bf16 v2bf;
typedef __attribute__((ext_vector_type(8)))  float  v8f;

// Pack two f32 into one dword of two bf16 (v_cvt_pk_bf16_f32)
__device__ __forceinline__ uint32_t pack_bf16(float a, float b) {
    v2bf p = { (__bf16)a, (__bf16)b };
    return __builtin_bit_cast(uint32_t, p);
}

// Assemble a 16-element bf16 WMMA fragment from 8 packed dwords
__device__ __forceinline__ v16bf frag_from(uint4 lo, uint4 hi) {
    uint32_t d[8] = { lo.x, lo.y, lo.z, lo.w, hi.x, hi.y, hi.z, hi.w };
    return __builtin_bit_cast(v16bf, *(uint32_t(*)[8])d);
}

// Build A fragment (ISA 7.12.2 16-bit A 16x32 layout) from two contiguous
// 8-float runs of the lane's row: K = k0+8*half..+7 and K = k0+16+8*half..+7.
__device__ __forceinline__ v16bf a_frag(float4 a0, float4 a1, float4 a2, float4 a3) {
    uint4 lo, hi;
    lo.x = pack_bf16(a0.x, a0.y);  lo.y = pack_bf16(a0.z, a0.w);
    lo.z = pack_bf16(a1.x, a1.y);  lo.w = pack_bf16(a1.z, a1.w);
    hi.x = pack_bf16(a2.x, a2.y);  hi.y = pack_bf16(a2.z, a2.w);
    hi.z = pack_bf16(a3.x, a3.y);  hi.w = pack_bf16(a3.z, a3.w);
    return frag_from(lo, hi);
}

__device__ __forceinline__ float4 relu4(float4 a) {
    a.x = a.x > 0.f ? a.x : 0.f;  a.y = a.y > 0.f ? a.y : 0.f;
    a.z = a.z > 0.f ? a.z : 0.f;  a.w = a.w > 0.f ? a.w : 0.f;
    return a;
}

// ---------------------------------------------------------------------------
// GEMM1: h1[100000,64] = x[100000,256] @ W1[256,64]  (bf16 WMMA, f32 accum)
// One wave -> one 16x16 tile. blockIdx.y = 16-col tile (0..3).
// W slice pre-converted to packed-bf16 in LDS, layout [n][kpair], row pad +4.
// ---------------------------------------------------------------------------
#define KP1 128           // 256/2 K-pairs
#define WPITCH1 (KP1 + 4) // pad: lane n starts at bank 4n -> conflict-free b128
__global__ void __launch_bounds__(256)
gemm1_kernel(const float* __restrict__ x, const float* __restrict__ W1,
             float* __restrict__ h1) {
    __shared__ uint32_t sW[16 * WPITCH1];
    const int nt  = blockIdx.y;
    const int tid = threadIdx.x;
    for (int i = tid; i < 16 * KP1; i += 256) {
        const int n  = i >> 7;          // 0..15
        const int kp = i & (KP1 - 1);   // K-pair index
        const float w0 = W1[(2 * kp)     * 64 + nt * 16 + n];
        const float w1 = W1[(2 * kp + 1) * 64 + nt * 16 + n];
        sW[n * WPITCH1 + kp] = pack_bf16(w0, w1);
    }
    __syncthreads();

    const int wave = tid >> 5;
    const int lane = tid & 31;
    const int half = lane >> 4;          // 0 = lanes 0-15, 1 = lanes 16-31
    const int m    = lane & 15;
    const long mt  = (long)blockIdx.x * 8 + wave;
    if (mt >= N_TILES) return;

    const float4*   arow = (const float4*)(x + (mt * 16 + m) * 256);
    const uint32_t* wrow = &sW[m * WPITCH1];

    v8f acc = {};
#pragma unroll 2
    for (int k0 = 0; k0 < 256; k0 += 32) {
        const int f0 = (k0 + 8 * half) >> 2;          // float4 index of run 0
        const float4 a0 = arow[f0],     a1 = arow[f0 + 1];
        const float4 a2 = arow[f0 + 4], a3 = arow[f0 + 5];  // K+16 run
        const int kp0 = (k0 >> 1) + 4 * half;
        const uint4 b0 = *(const uint4*)&wrow[kp0];
        const uint4 b1 = *(const uint4*)&wrow[kp0 + 8];
        acc = __builtin_amdgcn_wmma_f32_16x16x32_bf16(
            false, a_frag(a0, a1, a2, a3), false, frag_from(b0, b1),
            (short)0, acc, false, false);
    }
    // C/D layout: VGPR r -> M = r + 8*half, N = lane&15
#pragma unroll
    for (int r = 0; r < 8; ++r) {
        const int mm = r + half * 8;
        h1[(mt * 16 + mm) * 64 + nt * 16 + m] = acc[r];
    }
}

// ---------------------------------------------------------------------------
// GEMM2: h2[100000,16] = relu(h1s[100000,64]) @ W2[64,16]
// ---------------------------------------------------------------------------
#define KP2 32            // 64/2 K-pairs
#define WPITCH2 (KP2 + 4)
__global__ void __launch_bounds__(256)
gemm2_kernel(const float* __restrict__ h1s, const float* __restrict__ W2,
             float* __restrict__ h2) {
    __shared__ uint32_t sW[16 * WPITCH2];
    const int tid = threadIdx.x;
    for (int i = tid; i < 16 * KP2; i += 256) {
        const int n  = i >> 5;
        const int kp = i & (KP2 - 1);
        sW[n * WPITCH2 + kp] = pack_bf16(W2[(2 * kp) * 16 + n],
                                         W2[(2 * kp + 1) * 16 + n]);
    }
    __syncthreads();

    const int wave = tid >> 5;
    const int lane = tid & 31;
    const int half = lane >> 4;
    const int m    = lane & 15;
    const long mt  = (long)blockIdx.x * 8 + wave;
    if (mt >= N_TILES) return;

    const float4*   arow = (const float4*)(h1s + (mt * 16 + m) * 64);
    const uint32_t* wrow = &sW[m * WPITCH2];

    v8f acc = {};
#pragma unroll
    for (int k0 = 0; k0 < 64; k0 += 32) {
        const int f0 = (k0 + 8 * half) >> 2;
        const float4 a0 = relu4(arow[f0]),     a1 = relu4(arow[f0 + 1]);
        const float4 a2 = relu4(arow[f0 + 4]), a3 = relu4(arow[f0 + 5]);
        const int kp0 = (k0 >> 1) + 4 * half;
        const uint4 b0 = *(const uint4*)&wrow[kp0];
        const uint4 b1 = *(const uint4*)&wrow[kp0 + 8];
        acc = __builtin_amdgcn_wmma_f32_16x16x32_bf16(
            false, a_frag(a0, a1, a2, a3), false, frag_from(b0, b1),
            (short)0, acc, false, false);
    }
#pragma unroll
    for (int r = 0; r < 8; ++r) {
        const int mm = r + half * 8;
        h2[(mt * 16 + mm) * 16 + m] = acc[r];
    }
}

// ---------------------------------------------------------------------------
// SpMM scatter: out[dst] += val * h[src]; F/4 float4 groups per edge.
// ---------------------------------------------------------------------------
template <int F>
__global__ void __launch_bounds__(256)
spmm_kernel(const long long* __restrict__ dst, const long long* __restrict__ src,
            const float* __restrict__ vals, const float* __restrict__ h,
            float* __restrict__ out, int nedges) {
    constexpr int PER = F / 4;
    const long tid   = (long)blockIdx.x * blockDim.x + threadIdx.x;
    const long total = (long)nedges * PER;
    if (tid >= total) return;
    const long e = tid / PER;
    const int  g = (int)(tid % PER);

    if (e + 2048 < nedges) {                  // global_prefetch_b8 on edge stream
        __builtin_prefetch(&src[e + 2048], 0, 0);
        __builtin_prefetch(&dst[e + 2048], 0, 0);
    }

    const long long s = src[e];
    const long long d = dst[e];
    const float     v = vals[e];
    const float4   hv = ((const float4*)(h + (long)s * F))[g];
    float* o = out + (long)d * F + g * 4;
    __hip_atomic_fetch_add(o + 0, v * hv.x, __ATOMIC_RELAXED, __HIP_MEMORY_SCOPE_AGENT);
    __hip_atomic_fetch_add(o + 1, v * hv.y, __ATOMIC_RELAXED, __HIP_MEMORY_SCOPE_AGENT);
    __hip_atomic_fetch_add(o + 2, v * hv.z, __ATOMIC_RELAXED, __HIP_MEMORY_SCOPE_AGENT);
    __hip_atomic_fetch_add(o + 3, v * hv.w, __ATOMIC_RELAXED, __HIP_MEMORY_SCOPE_AGENT);
}

__global__ void __launch_bounds__(256)
zero_kernel(float4* __restrict__ p, long n4) {
    const long i = (long)blockIdx.x * blockDim.x + threadIdx.x;
    if (i < n4) p[i] = make_float4(0.f, 0.f, 0.f, 0.f);
}

// ---------------------------------------------------------------------------
// Row-wise log_softmax over 16 classes. 16 lanes/row, wave32 = 2 rows/wave.
// ---------------------------------------------------------------------------
__global__ void __launch_bounds__(256)
lsm_kernel(const float* __restrict__ h2s, float* __restrict__ out) {
    const long tid = (long)blockIdx.x * blockDim.x + threadIdx.x;
    const long row = tid >> 4;
    const int  c   = (int)(tid & 15);
    if (row >= N_NODES) return;
    const float v = h2s[row * 16 + c];
    float m = v;
#pragma unroll
    for (int o = 8; o > 0; o >>= 1) m = fmaxf(m, __shfl_xor(m, o, 16));
    const float e = __expf(v - m);
    float s = e;
#pragma unroll
    for (int o = 8; o > 0; o >>= 1) s += __shfl_xor(s, o, 16);
    out[row * 16 + c] = (v - m) - __logf(s);
}

// ---------------------------------------------------------------------------
extern "C" void kernel_launch(void* const* d_in, const int* in_sizes, int n_in,
                              void* d_out, int out_size, void* d_ws, size_t ws_size,
                              hipStream_t stream) {
    const float*     x     = (const float*)d_in[0];        // [100000,256]
    const float*     W1    = (const float*)d_in[1];        // [256,64]
    const float*     W2    = (const float*)d_in[2];        // [64,16]
    const float*     evals = (const float*)d_in[3];        // [E]
    const long long* eidx  = (const long long*)d_in[4];    // [2,E] int64
    float*           out   = (float*)d_out;                // [100000,16]

    const int E = in_sizes[3];
    const long long* dst = eidx;       // edge_index[0]
    const long long* src = eidx + E;   // edge_index[1]

    float* h1  = (float*)d_ws;                     // [100000,64]
    float* h1s = h1  + (size_t)N_NODES * 64;       // [100000,64]
    float* h2  = h1s + (size_t)N_NODES * 64;       // [100000,16]
    float* h2s = h2  + (size_t)N_NODES * 16;       // [100000,16]

    const int gemm_blocks = (N_TILES + 7) / 8;     // 8 waves / block

    // zero scatter targets
    {
        const long n4a = (long)N_NODES * 64 / 4;
        zero_kernel<<<(int)((n4a + 255) / 256), 256, 0, stream>>>((float4*)h1s, n4a);
        const long n4b = (long)N_NODES * 16 / 4;
        zero_kernel<<<(int)((n4b + 255) / 256), 256, 0, stream>>>((float4*)h2s, n4b);
    }

    // layer 1: GEMM -> SpMM
    gemm1_kernel<<<dim3(gemm_blocks, 4), 256, 0, stream>>>(x, W1, h1);
    {
        const long total = (long)E * 16;           // 64 feats / 4 per thread
        spmm_kernel<64><<<(int)((total + 255) / 256), 256, 0, stream>>>(
            dst, src, evals, h1, h1s, E);
    }

    // layer 2: ReLU-fused GEMM -> SpMM
    gemm2_kernel<<<gemm_blocks, 256, 0, stream>>>(h1s, W2, h2);
    {
        const long total = (long)E * 4;            // 16 feats / 4 per thread
        spmm_kernel<16><<<(int)((total + 255) / 256), 256, 0, stream>>>(
            dst, src, evals, h2, h2s, E);
    }

    // log_softmax
    {
        const long total = (long)N_NODES * 16;
        lsm_kernel<<<(int)((total + 255) / 256), 256, 0, stream>>>(h2s, out);
    }
}
